// SDPAtt_24094766530777
// MI455X (gfx1250) — matrix operand order repaired and checked by reading, not compiled, not run on hardware
//
#include <hip/hip_runtime.h>
#include <hip/hip_bf16.h>
#include <stdint.h>

typedef __bf16 bf16;
typedef __attribute__((ext_vector_type(16))) __bf16 v16bf;
typedef __attribute__((ext_vector_type(8)))  float  v8f;
typedef __attribute__((ext_vector_type(4)))  int    v4i;

union BF16x8  { uint4 u;    bf16 h[8];  };
union BF16x16 { uint4 u[2]; v16bf v;    };

typedef __attribute__((address_space(1))) v4i g_v4i;   // global int4
typedef __attribute__((address_space(3))) v4i l_v4i;   // LDS int4

// ---- CDNA5 async global->LDS copy (ASYNCcnt path) ----------------------
static __device__ inline void async_b128(void* lds, const void* g) {
#if __has_builtin(__builtin_amdgcn_global_load_async_to_lds_b128)
  __builtin_amdgcn_global_load_async_to_lds_b128(
      (g_v4i*)(uintptr_t)g, (l_v4i*)(uint32_t)(uintptr_t)lds, 0, 0);
#else
  asm volatile("global_load_async_to_lds_b128 %0, %1, off"
               :: "v"((uint32_t)(uintptr_t)lds),
                  "v"((uint64_t)(uintptr_t)g)
               : "memory");
#endif
}

static __device__ inline void wait_async0() {
#if __has_builtin(__builtin_amdgcn_s_wait_asynccnt)
  __builtin_amdgcn_s_wait_asynccnt(0);
#else
  asm volatile("s_wait_asynccnt 0x0" ::: "memory");
#endif
}

static __device__ inline v8f wmma_bf16(const v16bf& a, const v16bf& b, v8f c) {
  return __builtin_amdgcn_wmma_f32_16x16x32_bf16(false, a, false, b, (short)0, c,
                                                 false, false);
}

static __device__ inline v8f zero8() {
  v8f z = {0.f, 0.f, 0.f, 0.f, 0.f, 0.f, 0.f, 0.f};
  return z;
}

// A fragment: 16x32 (MxK) bf16. lane&15 -> row; K halves per lane>>4.
static __device__ inline v16bf load_frag_a(const bf16* lds, int row0, int stride,
                                           int kbase, int lane) {
  int r = row0 + (lane & 15);
  int k = kbase + (lane >> 4) * 8;
  BF16x16 f;
  f.u[0] = *(const uint4*)(lds + r * stride + k);
  f.u[1] = *(const uint4*)(lds + r * stride + k + 16);
  return f.v;
}

// B fragment: 32x16 (KxN) bf16 stored LDS[n][k] (k contiguous).
static __device__ inline v16bf load_frag_b(const bf16* lds, int col0, int stride,
                                           int kbase, int lane) {
  int c = col0 + (lane & 15);
  int k = kbase + (lane >> 4) * 8;
  BF16x16 f;
  f.u[0] = *(const uint4*)(lds + c * stride + k);
  f.u[1] = *(const uint4*)(lds + c * stride + k + 16);
  return f.v;
}

static __device__ inline uint4 cvt8(float4 a, float4 b) {
  BF16x8 p;
  p.h[0]=(bf16)a.x; p.h[1]=(bf16)a.y; p.h[2]=(bf16)a.z; p.h[3]=(bf16)a.w;
  p.h[4]=(bf16)b.x; p.h[5]=(bf16)b.y; p.h[6]=(bf16)b.z; p.h[7]=(bf16)b.w;
  return p.u;
}

// ---------------------------------------------------------------------------
#define BATCH 8
#define NQ 1024
#define NK 1024
#define CDIM 1024
#define NHEAD 16
#define HD 64
#define MROWS (BATCH * NQ)   // 8192

#define PBM 128
#define PBN 128
#define PBK 32
#define A_STRIDE 40          // padded LDS stride (bf16), 16B rows, bank-staggered
#define B_STRIDE 40

#define S_STRIDE 1032        // f32, 16B rows, bank-staggered
#define KV_STRIDE 72         // bf16, 16B rows, bank-staggered

// copy 128x32 bf16 tile (src row stride = src_stride elems) into LDS (stride 40)
static __device__ inline void stage_tile128x32(bf16* lds, const bf16* src,
                                               int src_stride, int t) {
  int r = t >> 1, c = (t & 1) * 16;
  const bf16* s = src + (size_t)r * src_stride + c;
  bf16* d = lds + r * A_STRIDE + c;
  async_b128(d, s);
  async_b128(d + 8, s + 8);
}

// copy 64x64 bf16 tile into LDS (stride 72)
static __device__ inline void stage_tile64x64(bf16* lds, const bf16* src,
                                              int src_stride, int t) {
  int r = t >> 2, c = (t & 3) * 16;
  const bf16* s = src + (size_t)r * src_stride + c;
  bf16* d = lds + r * KV_STRIDE + c;
  async_b128(d, s);
  async_b128(d + 8, s + 8);
}

// ---------------------------------------------------------------------------
// Prep 1: f32 -> bf16 conversion of Q/K/V activations (once).
__global__ __launch_bounds__(256) void SDP_cvt_bf16(
    const float* __restrict__ q, const float* __restrict__ k,
    const float* __restrict__ v, bf16* __restrict__ oq, bf16* __restrict__ ok2,
    bf16* __restrict__ ov) {
  const int z = blockIdx.z;
  const float* src = (z == 0) ? q : (z == 1) ? k : v;
  bf16* dst = (z == 0) ? oq : (z == 1) ? ok2 : ov;
  size_t i = ((size_t)blockIdx.x * 256 + threadIdx.x) * 8;
  const float4* s4 = (const float4*)(src + i);
  float4 a = s4[0], b = s4[1];
  *(uint4*)(dst + i) = cvt8(a, b);
}

// Prep 2: W[k][n] f32 -> WT[n][k] bf16 (LDS-tiled transpose), once per weight.
__global__ __launch_bounds__(256) void SDP_transpose_w(
    const float* __restrict__ Wq, const float* __restrict__ Wk,
    const float* __restrict__ Wv, const float* __restrict__ Wo,
    bf16* __restrict__ Tq, bf16* __restrict__ Tk, bf16* __restrict__ Tv,
    bf16* __restrict__ To) {
  __shared__ bf16 tile[64 * KV_STRIDE];
  const int z = blockIdx.z;
  const float* W = (z == 0) ? Wq : (z == 1) ? Wk : (z == 2) ? Wv : Wo;
  bf16* T = (z == 0) ? Tq : (z == 1) ? Tk : (z == 2) ? Tv : To;
  const int k0 = blockIdx.y * 64, n0 = blockIdx.x * 64;
  const int t = threadIdx.x;
  {
    int r = t >> 2, c = (t & 3) * 16;
    const float4* s = (const float4*)(W + (size_t)(k0 + r) * CDIM + n0 + c);
    float4 f0 = s[0], f1 = s[1], f2 = s[2], f3 = s[3];
    *(uint4*)&tile[r * KV_STRIDE + c]     = cvt8(f0, f1);
    *(uint4*)&tile[r * KV_STRIDE + c + 8] = cvt8(f2, f3);
  }
  __syncthreads();
  {
    int nr = t >> 2, kc = (t & 3) * 16;
    BF16x16 p;
#pragma unroll
    for (int i = 0; i < 16; ++i) p.v[i] = tile[(kc + i) * KV_STRIDE + nr];
    uint4* d = (uint4*)(T + (size_t)(n0 + nr) * CDIM + k0 + kc);
    d[0] = p.u[0];
    d[1] = p.u[1];
  }
}

// ---------------------------------------------------------------------------
// Kernel 1: QKV projection, async double-buffered bf16 GEMM.
// q -> [b][h][n][d] * 1/8 ; k -> [b][h][n][d] ; v -> [b][h][d][n] (transposed)
__global__ __launch_bounds__(256) void SDP_proj_qkv(
    const bf16* __restrict__ Xq, const bf16* __restrict__ Xk,
    const bf16* __restrict__ Xv, const bf16* __restrict__ Tq,
    const bf16* __restrict__ Tk, const bf16* __restrict__ Tv,
    const float* __restrict__ bq, const float* __restrict__ bk,
    const float* __restrict__ bv, bf16* __restrict__ oq,
    bf16* __restrict__ ok2, bf16* __restrict__ ov) {
  __shared__ bf16 As[2][PBM * A_STRIDE];
  __shared__ bf16 Bs[2][PBN * B_STRIDE];

  const int z = blockIdx.z;
  const bf16* X = (z == 0) ? Xq : (z == 1) ? Xk : Xv;
  const bf16* T = (z == 0) ? Tq : (z == 1) ? Tk : Tv;
  const float* bias = (z == 0) ? bq : (z == 1) ? bk : bv;
  bf16* out = (z == 0) ? oq : (z == 1) ? ok2 : ov;
  const float scale = (z == 0) ? 0.125f : 1.0f;

  const int m0 = blockIdx.y * PBM, n0 = blockIdx.x * PBN;
  const int t = threadIdx.x, w = t >> 5, lane = t & 31;
  const int mb = (w >> 1) * 32, nb = (w & 1) * 64;

  v8f acc[2][4];
  for (int i = 0; i < 2; ++i)
    for (int j = 0; j < 4; ++j) acc[i][j] = zero8();

  stage_tile128x32(As[0], X + (size_t)m0 * CDIM, CDIM, t);
  stage_tile128x32(Bs[0], T + (size_t)n0 * CDIM, CDIM, t);
  wait_async0();
  __syncthreads();

  const int NT = CDIM / PBK;  // 32
  for (int i = 0; i < NT; ++i) {
    int nxt = i + 1;
    if (nxt < NT) {
      stage_tile128x32(As[nxt & 1], X + (size_t)m0 * CDIM + nxt * PBK, CDIM, t);
      stage_tile128x32(Bs[nxt & 1], T + (size_t)n0 * CDIM + nxt * PBK, CDIM, t);
    }
    const bf16* a_ = As[i & 1];
    const bf16* b_ = Bs[i & 1];
    v16bf af[2], bfr[4];
#pragma unroll
    for (int rt = 0; rt < 2; ++rt) af[rt]  = load_frag_a(a_, mb + rt * 16, A_STRIDE, 0, lane);
#pragma unroll
    for (int ct = 0; ct < 4; ++ct) bfr[ct] = load_frag_b(b_, nb + ct * 16, B_STRIDE, 0, lane);
#pragma unroll
    for (int rt = 0; rt < 2; ++rt)
#pragma unroll
      for (int ct = 0; ct < 4; ++ct) acc[rt][ct] = wmma_bf16(af[rt], bfr[ct], acc[rt][ct]);
    if (nxt < NT) {
      wait_async0();
      __syncthreads();
    }
  }

  const int hl = lane >> 4, nl = lane & 15;
#pragma unroll
  for (int rt = 0; rt < 2; ++rt)
#pragma unroll
    for (int ct = 0; ct < 4; ++ct)
#pragma unroll
      for (int i = 0; i < 8; ++i) {
        int row = m0 + mb + rt * 16 + i + hl * 8;
        int col = n0 + nb + ct * 16 + nl;
        float val = (acc[rt][ct][i] + bias[col]) * scale;
        int b = row >> 10, n = row & 1023;
        int hh = col >> 6, d = col & 63;
        size_t idx = (z == 2)
                         ? (((size_t)b * NHEAD + hh) * HD + d) * NK + n
                         : (((size_t)b * NHEAD + hh) * NQ + n) * HD + d;
        out[idx] = (bf16)val;
      }
}

// ---------------------------------------------------------------------------
// Kernel 2: attention for one (b,h,64-query slab); async K/V double-buffering.
__global__ __launch_bounds__(256) void SDP_attention(
    const bf16* __restrict__ qb, const bf16* __restrict__ kbm,
    const bf16* __restrict__ vbm, const float* __restrict__ AW,
    const unsigned char* __restrict__ AM, bf16* __restrict__ out) {
  extern __shared__ char smem[];
  float* S = (float*)smem;                              // 64 x S_STRIDE f32
  bf16* qs = (bf16*)(smem + 64 * S_STRIDE * 4);         // 64 x KV_STRIDE
  bf16* kvb[2] = {qs + 64 * KV_STRIDE, qs + 2 * 64 * KV_STRIDE};

  const int bh = blockIdx.y;
  const int q0 = blockIdx.x * 64;
  const int t = threadIdx.x, w = t >> 5, lane = t & 31;
  const int b = bh >> 4, h = bh & 15;
  const size_t headoff = (size_t)bh * NK * HD;
  const bf16* kbase = kbm + headoff;   // [key][d], stride HD
  const bf16* vbase = vbm + headoff;   // [d][key], stride NK

  stage_tile64x64(qs, qb + headoff + (size_t)q0 * HD, HD, t);
  stage_tile64x64(kvb[0], kbase, HD, t);
  wait_async0();
  __syncthreads();

  const int mb = (w >> 1) * 16;
  const int nb = (w & 1) * 32;
  const int hl = lane >> 4, nl = lane & 15;

  // ---- phase 1: scores ----
  for (int kblk = 0; kblk < NK / 64; ++kblk) {
    int nxt = kblk + 1;
    if (nxt < NK / 64)
      stage_tile64x64(kvb[nxt & 1], kbase + (size_t)nxt * 64 * HD, HD, t);

    const bf16* kt = kvb[kblk & 1];
    v8f acc[2] = {zero8(), zero8()};
#pragma unroll
    for (int ks = 0; ks < HD; ks += 32) {
      v16bf af = load_frag_a(qs, mb, KV_STRIDE, ks, lane);
#pragma unroll
      for (int ct = 0; ct < 2; ++ct) {
        v16bf bfg = load_frag_b(kt, nb + ct * 16, KV_STRIDE, ks, lane);
        acc[ct] = wmma_bf16(af, bfg, acc[ct]);
      }
    }
#pragma unroll
    for (int ct = 0; ct < 2; ++ct)
#pragma unroll
      for (int i = 0; i < 8; ++i) {
        int ql = mb + i + hl * 8;
        int kg = kblk * 64 + nb + ct * 16 + nl;
        size_t idx = ((size_t)bh * NQ + (q0 + ql)) * NK + kg;
        float s = acc[ct][i] * AW[idx];
        if (AM[idx]) s = -3.0e38f;
        S[ql * S_STRIDE + kg] = s;
      }
    if (nxt < NK / 64) {
      wait_async0();
      __syncthreads();
    }
  }
  __syncthreads();

  // prefetch first V block; copy hides under softmax
  stage_tile64x64(kvb[0], vbase, NK, t);

  // ---- softmax: wave w owns rows w*8 .. w*8+7 ----
  for (int r = w * 8; r < w * 8 + 8; ++r) {
    float mx = -3.4e38f;
    for (int j = lane; j < NK; j += 32) mx = fmaxf(mx, S[r * S_STRIDE + j]);
#pragma unroll
    for (int off = 16; off; off >>= 1) mx = fmaxf(mx, __shfl_xor(mx, off, 32));
    float sum = 0.f;
    for (int j = lane; j < NK; j += 32) {
      float e = __expf(S[r * S_STRIDE + j] - mx);
      S[r * S_STRIDE + j] = e;
      sum += e;
    }
#pragma unroll
    for (int off = 16; off; off >>= 1) sum += __shfl_xor(sum, off, 32);
    float inv = 1.0f / sum;
    for (int j = lane; j < NK; j += 32) S[r * S_STRIDE + j] *= inv;
  }
  wait_async0();
  __syncthreads();

  // ---- phase 2: O = P @ V ----
  v8f o[2] = {zero8(), zero8()};
  for (int kblk = 0; kblk < NK / 64; ++kblk) {
    int nxt = kblk + 1;
    if (nxt < NK / 64)
      stage_tile64x64(kvb[nxt & 1], vbase + nxt * 64, NK, t);

    const bf16* vt = kvb[kblk & 1];
#pragma unroll
    for (int ks = 0; ks < 64; ks += 32) {
      int r = mb + (lane & 15);
      int base = kblk * 64 + ks + (lane >> 4) * 8;
      v16bf af;
#pragma unroll
      for (int i = 0; i < 8; ++i) af[i]     = (bf16)S[r * S_STRIDE + base + i];
#pragma unroll
      for (int i = 0; i < 8; ++i) af[8 + i] = (bf16)S[r * S_STRIDE + base + 16 + i];
#pragma unroll
      for (int ct = 0; ct < 2; ++ct) {
        v16bf bfg = load_frag_b(vt, nb + ct * 16, KV_STRIDE, ks, lane);
        o[ct] = wmma_bf16(af, bfg, o[ct]);
      }
    }
    if (nxt < NK / 64) {
      wait_async0();
      __syncthreads();
    }
  }
#pragma unroll
  for (int ct = 0; ct < 2; ++ct)
#pragma unroll
    for (int i = 0; i < 8; ++i) {
      int ql = mb + i + hl * 8;
      int d = nb + ct * 16 + nl;
      out[((size_t)b * NQ + q0 + ql) * (NHEAD * HD) + h * HD + d] = (bf16)o[ct][i];
    }
}

// ---------------------------------------------------------------------------
// Kernel 3: output projection f32 = attout(bf16)[8192x1024] @ WoT + bo
__global__ __launch_bounds__(256) void SDP_outproj(
    const bf16* __restrict__ A, const bf16* __restrict__ To,
    const float* __restrict__ bo, float* __restrict__ out) {
  __shared__ bf16 As[2][PBM * A_STRIDE];
  __shared__ bf16 Bs[2][PBN * B_STRIDE];

  const int m0 = blockIdx.y * PBM, n0 = blockIdx.x * PBN;
  const int t = threadIdx.x, w = t >> 5, lane = t & 31;
  const int mb = (w >> 1) * 32, nb = (w & 1) * 64;

  v8f acc[2][4];
  for (int i = 0; i < 2; ++i)
    for (int j = 0; j < 4; ++j) acc[i][j] = zero8();

  stage_tile128x32(As[0], A + (size_t)m0 * CDIM, CDIM, t);
  stage_tile128x32(Bs[0], To + (size_t)n0 * CDIM, CDIM, t);
  wait_async0();
  __syncthreads();

  const int NT = CDIM / PBK;
  for (int i = 0; i < NT; ++i) {
    int nxt = i + 1;
    if (nxt < NT) {
      stage_tile128x32(As[nxt & 1], A + (size_t)m0 * CDIM + nxt * PBK, CDIM, t);
      stage_tile128x32(Bs[nxt & 1], To + (size_t)n0 * CDIM + nxt * PBK, CDIM, t);
    }
    const bf16* a_ = As[i & 1];
    const bf16* b_ = Bs[i & 1];
    v16bf af[2], bfr[4];
#pragma unroll
    for (int rt = 0; rt < 2; ++rt) af[rt]  = load_frag_a(a_, mb + rt * 16, A_STRIDE, 0, lane);
#pragma unroll
    for (int ct = 0; ct < 4; ++ct) bfr[ct] = load_frag_b(b_, nb + ct * 16, B_STRIDE, 0, lane);
#pragma unroll
    for (int rt = 0; rt < 2; ++rt)
#pragma unroll
      for (int ct = 0; ct < 4; ++ct) acc[rt][ct] = wmma_bf16(af[rt], bfr[ct], acc[rt][ct]);
    if (nxt < NT) {
      wait_async0();
      __syncthreads();
    }
  }

  const int hl = lane >> 4, nl = lane & 15;
#pragma unroll
  for (int rt = 0; rt < 2; ++rt)
#pragma unroll
    for (int ct = 0; ct < 4; ++ct)
#pragma unroll
      for (int i = 0; i < 8; ++i) {
        int row = m0 + mb + rt * 16 + i + hl * 8;
        int col = n0 + nb + ct * 16 + nl;
        out[(size_t)row * CDIM + col] = acc[rt][ct][i] + bo[col];
      }
}

// ---------------------------------------------------------------------------
extern "C" void kernel_launch(void* const* d_in, const int* in_sizes, int n_in,
                              void* d_out, int out_size, void* d_ws,
                              size_t ws_size, hipStream_t stream) {
  (void)in_sizes; (void)n_in; (void)out_size; (void)ws_size;

  const float* queries = (const float*)d_in[0];
  const float* keys    = (const float*)d_in[1];
  const float* values  = (const float*)d_in[2];
  const unsigned char* amask = (const unsigned char*)d_in[3];  // jnp.bool_ = 1B
  const float* aw = (const float*)d_in[4];
  const float* Wq = (const float*)d_in[5];
  const float* bq = (const float*)d_in[6];
  const float* Wk = (const float*)d_in[7];
  const float* bk = (const float*)d_in[8];
  const float* Wv = (const float*)d_in[9];
  const float* bv = (const float*)d_in[10];
  const float* Wo = (const float*)d_in[11];
  const float* bo = (const float*)d_in[12];
  float* out = (float*)d_out;

  // workspace layout (bytes):
  //  [0,48M)  Xq|Xk|Xv bf16 activations   (attout later aliases Xq region)
  //  [48M,56M) WTq|WTk|WTv|WTo bf16 transposed weights (2MB each)
  //  [56M,104M) qbf|kbf|vbf bf16 projected tensors (16MB each)
  const size_t XSEG = (size_t)MROWS * CDIM * sizeof(bf16);  // 16 MiB
  const size_t WSEG = (size_t)CDIM * CDIM * sizeof(bf16);   // 2 MiB
  char* ws = (char*)d_ws;
  bf16* Xq = (bf16*)(ws + 0 * XSEG);
  bf16* Xk = (bf16*)(ws + 1 * XSEG);
  bf16* Xv = (bf16*)(ws + 2 * XSEG);
  bf16* Tq = (bf16*)(ws + 3 * XSEG + 0 * WSEG);
  bf16* Tk = (bf16*)(ws + 3 * XSEG + 1 * WSEG);
  bf16* Tv = (bf16*)(ws + 3 * XSEG + 2 * WSEG);
  bf16* To = (bf16*)(ws + 3 * XSEG + 3 * WSEG);
  bf16* qbf = (bf16*)(ws + 3 * XSEG + 4 * WSEG);
  bf16* kbf = qbf + (size_t)MROWS * CDIM;
  bf16* vbf = kbf + (size_t)MROWS * CDIM;
  bf16* attout = Xq;  // Xq dead after projection kernel

  // prep
  dim3 gCvt((unsigned)(((size_t)MROWS * CDIM / 8) / 256), 1, 3);
  SDP_cvt_bf16<<<gCvt, 256, 0, stream>>>(queries, keys, values, Xq, Xk, Xv);
  dim3 gTr(CDIM / 64, CDIM / 64, 4);
  SDP_transpose_w<<<gTr, 256, 0, stream>>>(Wq, Wk, Wv, Wo, Tq, Tk, Tv, To);

  // 1) QKV projections
  dim3 gProj(CDIM / PBN, MROWS / PBM, 3);
  SDP_proj_qkv<<<gProj, 256, 0, stream>>>(Xq, Xk, Xv, Tq, Tk, Tv, bq, bk, bv,
                                          qbf, kbf, vbf);

  // 2) attention
  size_t smem = (size_t)64 * S_STRIDE * sizeof(float) +
                (size_t)3 * 64 * KV_STRIDE * sizeof(bf16);
  dim3 gAtt(NQ / 64, BATCH * NHEAD);
  SDP_attention<<<gAtt, 256, smem, stream>>>(qbf, kbf, vbf, aw, amask, attout);

  // 3) output projection
  dim3 gOut(CDIM / PBN, MROWS / PBM);
  SDP_outproj<<<gOut, 256, 0, stream>>>(attout, To, bo, out);
}
